// Attention_17815524343972
// MI455X (gfx1250) — matrix operand order repaired and checked by reading, not compiled, not run on hardware
//
#include <hip/hip_runtime.h>
#include <hip/hip_bf16.h>
#include <stdint.h>

// CDNA5 / gfx1250, wave32, WMMA f32 = f16 x f16 + f32 (16x16x32).
typedef __attribute__((ext_vector_type(16))) _Float16 v16h;
typedef __attribute__((ext_vector_type(8)))  float    v8f;

#define DIN 1024
#define DH  256
#define MAXW 33
#define NTW 4          // N-tiles (16 cols each) per wave -> 16x64 output strip

#define WMMA_F16(a, b, c) \
  __builtin_amdgcn_wmma_f32_16x16x32_f16(false, (a), false, (b), (short)0, (c), false, false)

// ---------------------------------------------------------------------------
// Pack f32 row-major [R x K] into A-fragment tiles (16x32), ISA 7.12.2 layout.
// Tile (rt,kt) id = rt*(K/32)+kt, 256 dwords/tile, dword idx = lane*8 + v.
// lane: m = lane&15, half = lane>>4 ; VGPR v holds K = (v&4)<<2 | half*8 | (v&3)*2, +1
// ---------------------------------------------------------------------------
__global__ __launch_bounds__(256) void pack_a_frag(const float* __restrict__ src,
                                                   uint32_t* __restrict__ dst,
                                                   int R, int K) {
  int tid = blockIdx.x * blockDim.x + threadIdx.x;
  int total = (R * K) >> 1;
  if (tid >= total) return;
  int tile = tid >> 8;
  int r8   = tid & 255;
  int lane = r8 >> 3;
  int v    = r8 & 7;
  int ktiles = K >> 5;
  int rt = tile / ktiles;
  int kt = tile - rt * ktiles;
  int row  = (rt << 4) + (lane & 15);
  int half = lane >> 4;
  int k0 = (kt << 5) + ((v & 4) << 2) + (half << 3) + ((v & 3) << 1);
  union { _Float16 h[2]; uint32_t u; } cvt;
  cvt.h[0] = (_Float16)src[(size_t)row * K + k0];
  cvt.h[1] = (_Float16)src[(size_t)row * K + k0 + 1];
  dst[tid] = cvt.u;
}

// ---------------------------------------------------------------------------
// Pack weight W [N x K] row-major into B-fragment tiles for B = W^T (K x N).
// B frag layout (ISA sparse 64x16 table halved for K=32):
//   lane: n = lane&15, half = lane>>4 ; VGPR v holds K = (v&4)<<2 | half*8 | (v&3)*2, +1
// Tile order groups a 64-col strip's 4 tiles contiguously per K-step:
//   id = (ng*ktn + kt)*4 + j  -> a wave's 4 B frags per K-step are one 4KB block.
// ---------------------------------------------------------------------------
__global__ __launch_bounds__(256) void pack_b_frag(const float* __restrict__ W,
                                                   uint32_t* __restrict__ dst,
                                                   int N, int K) {
  int tid = blockIdx.x * blockDim.x + threadIdx.x;
  int total = (N * K) >> 1;
  if (tid >= total) return;
  int tileIdx = tid >> 8;
  int r8   = tid & 255;
  int lane = r8 >> 3;
  int v    = r8 & 7;
  int ktn  = K >> 5;
  int j    = tileIdx & 3;
  int tmp  = tileIdx >> 2;
  int kt   = tmp % ktn;
  int ng   = tmp / ktn;
  int n    = ((ng << 2) + j) * 16 + (lane & 15);
  int half = lane >> 4;
  int k0 = (kt << 5) + ((v & 4) << 2) + (half << 3) + ((v & 3) << 1);
  union { _Float16 h[2]; uint32_t u; } cvt;
  cvt.h[0] = (_Float16)W[(size_t)n * K + k0];
  cvt.h[1] = (_Float16)W[(size_t)n * K + k0 + 1];
  dst[tid] = cvt.u;
}

// ---------------------------------------------------------------------------
// GEMM: C[M x N] = act(A @ B + bias). One wave per 16x64 output strip with
// explicit ping-pong double buffering. REQUIRES ktn = K/32 even and >= 2
// (true here: K = 1024 or 256); the final two phases are peeled so the
// steady-state loop body has no conditionals.
// ---------------------------------------------------------------------------
__global__ __launch_bounds__(256) void gemm_wmma_f16(const uint32_t* __restrict__ Apack,
                                                     const uint32_t* __restrict__ Bpack,
                                                     const float* __restrict__ bias,
                                                     float* __restrict__ C,
                                                     int M, int N, int K, int relu) {
  int lane = threadIdx.x & 31;
  int wid  = (blockIdx.x * blockDim.x + threadIdx.x) >> 5;
  int ngrp = N >> 6;                 // 64-column groups
  int mt   = wid / ngrp;
  int ng   = wid - mt * ngrp;
  if (mt >= (M >> 4)) return;
  int ktn = K >> 5;
  const uint32_t* aPtr = Apack + (((size_t)mt * ktn) << 8)  + (lane << 3);
  const uint32_t* bPtr = Bpack + (((size_t)ng * ktn) << 10) + (lane << 3);

  v8f acc0 = {}, acc1 = {}, acc2 = {}, acc3 = {};

  // phase X / phase Y fragment sets (both live simultaneously)
  v16h aX, bX0, bX1, bX2, bX3;
  v16h aY, bY0, bY1, bY2, bY3;

  aX  = *(const v16h*)(aPtr);
  bX0 = *(const v16h*)(bPtr);
  bX1 = *(const v16h*)(bPtr + 256);
  bX2 = *(const v16h*)(bPtr + 512);
  bX3 = *(const v16h*)(bPtr + 768);

  // steady state: phases kt (X) and kt+1 (Y); preload kt+2 into X.
  for (int kt = 0; kt + 2 < ktn; kt += 2) {
    __builtin_prefetch(aPtr + 512, 0, 1);    // global_prefetch_b8, phase kt+2
    __builtin_prefetch(bPtr + 2048, 0, 1);
    aY  = *(const v16h*)(aPtr + 256);        // loads for phase kt+1
    bY0 = *(const v16h*)(bPtr + 1024);
    bY1 = *(const v16h*)(bPtr + 1280);
    bY2 = *(const v16h*)(bPtr + 1536);
    bY3 = *(const v16h*)(bPtr + 1792);

    acc0 = WMMA_F16(aX, bX0, acc0);
    acc1 = WMMA_F16(aX, bX1, acc1);
    acc2 = WMMA_F16(aX, bX2, acc2);
    acc3 = WMMA_F16(aX, bX3, acc3);

    aPtr += 512;
    bPtr += 2048;
    aX  = *(const v16h*)(aPtr);              // loads for phase kt+2
    bX0 = *(const v16h*)(bPtr);
    bX1 = *(const v16h*)(bPtr + 256);
    bX2 = *(const v16h*)(bPtr + 512);
    bX3 = *(const v16h*)(bPtr + 768);

    acc0 = WMMA_F16(aY, bY0, acc0);
    acc1 = WMMA_F16(aY, bY1, acc1);
    acc2 = WMMA_F16(aY, bY2, acc2);
    acc3 = WMMA_F16(aY, bY3, acc3);
  }

  // epilogue: phases ktn-2 (X, already loaded) and ktn-1 (Y)
  aY  = *(const v16h*)(aPtr + 256);
  bY0 = *(const v16h*)(bPtr + 1024);
  bY1 = *(const v16h*)(bPtr + 1280);
  bY2 = *(const v16h*)(bPtr + 1536);
  bY3 = *(const v16h*)(bPtr + 1792);

  acc0 = WMMA_F16(aX, bX0, acc0);
  acc1 = WMMA_F16(aX, bX1, acc1);
  acc2 = WMMA_F16(aX, bX2, acc2);
  acc3 = WMMA_F16(aX, bX3, acc3);

  acc0 = WMMA_F16(aY, bY0, acc0);
  acc1 = WMMA_F16(aY, bY1, acc1);
  acc2 = WMMA_F16(aY, bY2, acc2);
  acc3 = WMMA_F16(aY, bY3, acc3);

  // C/D layout: lane<16 -> rows 0..7, lane>=16 -> rows 8..15; col = lane&15
  int half    = lane >> 4;
  int rowBase = (mt << 4) + (half << 3);
  v8f accs[NTW] = {acc0, acc1, acc2, acc3};
  #pragma unroll
  for (int j = 0; j < NTW; ++j) {
    int n = ((ng * NTW + j) << 4) + (lane & 15);
    float bv = bias[n];
    #pragma unroll
    for (int r = 0; r < 8; ++r) {
      float val = accs[j][r] + bv;
      if (relu) val = fmaxf(val, 0.f);
      C[(size_t)(rowBase + r) * N + n] = val;
    }
  }
}

// ---------------------------------------------------------------------------
// Banded "softmax attention" + blend. One block (256 thr) per row i.
// Only the width band of C = theta @ phi^T is observable after the mask, so
// compute just c_d = theta[i].phi[i+d]; max over C*M includes off-band zeros.
// A = M/rowsum(M) + exp(c-max)*valid / sum ; Ah[i] = sum_d A_d * h[i+d].
// ---------------------------------------------------------------------------
__global__ __launch_bounds__(256) void banded_gcn(const float* __restrict__ theta,
                                                  const float* __restrict__ phi,
                                                  const float* __restrict__ h,
                                                  float* __restrict__ Ah,
                                                  const int* __restrict__ widthPtr,
                                                  int T) {
  __shared__ float red[256];
  __shared__ float wsh[MAXW];
  int i = blockIdx.x;
  int j = threadIdx.x;
  int width = *widthPtr;
  if (width < 1) width = 1;
  if (width > MAXW) width = MAXW;
  int dmin = -(width / 2);  // floor(-(width-1)/2)

  float th = theta[(size_t)i * DH + j];
  for (int d = 0; d < width; ++d) {
    int jj = i + dmin + d;
    float p = 0.f;
    if (jj >= 0 && jj < T) p = th * phi[(size_t)jj * DH + j];
    red[j] = p;
    __syncthreads();
    for (int s = 128; s > 0; s >>= 1) {
      if (j < s) red[j] += red[j + s];
      __syncthreads();
    }
    if (j == 0) wsh[d] = red[0];
    __syncthreads();
  }
  if (j == 0) {
    float maxv = 0.f;  // off-band zeros of C*M participate in the row max
    int nvalid = 0;
    for (int d = 0; d < width; ++d) {
      int jj = i + dmin + d;
      if (jj >= 0 && jj < T) { nvalid++; maxv = fmaxf(maxv, wsh[d]); }
    }
    float e[MAXW];
    float sum = 0.f;
    for (int d = 0; d < width; ++d) {
      int jj = i + dmin + d;
      e[d] = (jj >= 0 && jj < T) ? expf(wsh[d] - maxv) : 0.f;
      sum += e[d];
    }
    float band = 1.f / (float)nvalid;
    for (int d = 0; d < width; ++d) {
      int jj = i + dmin + d;
      wsh[d] = (jj >= 0 && jj < T) ? (band + e[d] / sum) : 0.f;
    }
  }
  __syncthreads();
  float acc = 0.f;
  for (int d = 0; d < width; ++d) {
    int jj = i + dmin + d;
    if (jj >= 0 && jj < T) acc += wsh[d] * h[(size_t)jj * DH + j];
  }
  Ah[(size_t)i * DH + j] = acc;
}

// ---------------------------------------------------------------------------
// Head: y = h @ Wfc^T + bfc ; mask = sigmoid(y)+eps ; inv = 1/mask.
// ---------------------------------------------------------------------------
__global__ __launch_bounds__(256) void head_kernel(const float* __restrict__ h,
                                                   const float* __restrict__ Wfc,
                                                   const float* __restrict__ bfc,
                                                   float* __restrict__ mask,
                                                   float* __restrict__ inv) {
  __shared__ float red[256];
  int i = blockIdx.x;
  int j = threadIdx.x;
  red[j] = h[(size_t)i * DH + j] * Wfc[j];
  __syncthreads();
  for (int s = 128; s > 0; s >>= 1) {
    if (j < s) red[j] += red[j + s];
    __syncthreads();
  }
  if (j == 0) {
    float y = red[0] + bfc[0];
    float m = 1.f / (1.f + expf(-y)) + 1e-5f;
    mask[i] = m;
    inv[i]  = 1.f / m;
  }
}

// ---------------------------------------------------------------------------
extern "C" void kernel_launch(void* const* d_in, const int* in_sizes, int n_in,
                              void* d_out, int out_size, void* d_ws, size_t ws_size,
                              hipStream_t stream) {
  const float* x     = (const float*)d_in[0];
  const float* Wc    = (const float*)d_in[1];
  const float* bc    = (const float*)d_in[2];
  const float* Wt    = (const float*)d_in[3];
  const float* bt    = (const float*)d_in[4];
  const float* Wp    = (const float*)d_in[5];
  const float* bp    = (const float*)d_in[6];
  const float* Wd    = (const float*)d_in[7];
  const float* bd    = (const float*)d_in[8];
  const float* Wfc   = (const float*)d_in[9];
  const float* bfc   = (const float*)d_in[10];
  const int*   width = (const int*)d_in[11];

  const int T = in_sizes[0] / DIN;  // 4096

  // --- workspace layout (~29 MB) ---
  float* h     = (float*)d_ws;
  float* theta = h     + (size_t)T * DH;
  float* phi   = theta + (size_t)T * DH;
  float* Ah    = phi   + (size_t)T * DH;
  uint32_t* ApackX  = (uint32_t*)(Ah + (size_t)T * DH);
  uint32_t* ApackH  = ApackX  + ((size_t)T * DIN >> 1);
  uint32_t* ApackAh = ApackH  + ((size_t)T * DH  >> 1);
  uint32_t* BpackC  = ApackAh + ((size_t)T * DH  >> 1);
  uint32_t* BpackT  = BpackC  + ((size_t)DH * DIN >> 1);
  uint32_t* BpackP  = BpackT  + ((size_t)DH * DH  >> 1);
  uint32_t* BpackD  = BpackP  + ((size_t)DH * DH  >> 1);

  float* outMask = (float*)d_out;
  float* outInv  = outMask + T;
  float* outGcn  = outInv  + T;

  const int BS = 256;
  auto cdiv = [](int a, int b) { return (a + b - 1) / b; };

  // 1) pack operands into WMMA fragment order
  pack_a_frag<<<cdiv(T * DIN / 2, BS), BS, 0, stream>>>(x, ApackX, T, DIN);
  pack_b_frag<<<cdiv(DH * DIN / 2, BS), BS, 0, stream>>>(Wc, BpackC, DH, DIN);
  pack_b_frag<<<cdiv(DH * DH / 2, BS), BS, 0, stream>>>(Wt, BpackT, DH, DH);
  pack_b_frag<<<cdiv(DH * DH / 2, BS), BS, 0, stream>>>(Wp, BpackP, DH, DH);
  pack_b_frag<<<cdiv(DH * DH / 2, BS), BS, 0, stream>>>(Wd, BpackD, DH, DH);

  // waves: one per 16x64 strip
  int gemmWaves  = (T / 16) * (DH / 64);
  int gemmBlocks = cdiv(gemmWaves * 32, BS);

  // 2) h = relu(x @ Wc^T + bc)   [4096x1024x256]
  gemm_wmma_f16<<<gemmBlocks, BS, 0, stream>>>(ApackX, BpackC, bc, h,
                                               T, DH, DIN, /*relu=*/1);
  // 3) repack h as A-operand
  pack_a_frag<<<cdiv(T * DH / 2, BS), BS, 0, stream>>>(h, ApackH, T, DH);

  // 4) theta / phi  [4096x256x256]
  gemm_wmma_f16<<<gemmBlocks, BS, 0, stream>>>(ApackH, BpackT, bt, theta,
                                               T, DH, DH, 0);
  gemm_wmma_f16<<<gemmBlocks, BS, 0, stream>>>(ApackH, BpackP, bp, phi,
                                               T, DH, DH, 0);

  // 5) banded masked-softmax attention + A @ h
  banded_gcn<<<T, BS, 0, stream>>>(theta, phi, h, Ah, width, T);

  // 6) gcn_out = Ah @ Wd^T + bd -> d_out[2T:]
  pack_a_frag<<<cdiv(T * DH / 2, BS), BS, 0, stream>>>(Ah, ApackAh, T, DH);
  gemm_wmma_f16<<<gemmBlocks, BS, 0, stream>>>(ApackAh, BpackD, bd, outGcn,
                                               T, DH, DH, 0);

  // 7) head: mask / inverse_mask
  head_kernel<<<T, BS, 0, stream>>>(h, Wfc, bfc, outMask, outInv);
}
